// FMICLLoss_79663053406800
// MI455X (gfx1250) — compile-verified, matrix operand order checked by reading, not run on hardware
//
#include <hip/hip_runtime.h>
#include <hip/hip_bf16.h>

// ---------------------------------------------------------------------------
// FMICL loss for MI455X (gfx1250, wave32, WMMA).
//
// loss = -mean(s_pos) + ALPHA * ( sum_offdiag exp(-d2/2) / (N(N-1)) + EPS )
// since exp(log(g+eps)+1-1) == g+eps exactly.
//
// Pipeline:
//   1) normalize_kernel : row norms, f16 copy of normalized z1, sq[], s_pos[]
//   2) gram_kernel      : Z1n * Z1n^T via v_wmma_f32_16x16x32_f16; the Gram
//                         matrix is symmetric, so only the 528 upper-tri
//                         128x128 tile-blocks are computed (off-diag blocks
//                         weighted x2 — bitwise equal to computing both
//                         halves since c_mn == c_nm exactly). Fused
//                         exp-epilogue + per-block reduction; C never stored.
//   3) finalize_kernel  : deterministic reductions + scalar loss
// ---------------------------------------------------------------------------

typedef __attribute__((ext_vector_type(16))) _Float16 v16h;
typedef __attribute__((ext_vector_type(4)))  _Float16 v4h;
typedef __attribute__((ext_vector_type(8)))  float    v8f;

#define D_DIM   1024
#define SIGMA_INV2S2 0.5f   // 1/(2*sigma^2), sigma=1
#define ALPHA_C 64.0f
#define EPS_C   1e-8f

__device__ __forceinline__ float block_reduce_sum(float v, float* sm) {
  const int t = threadIdx.x;
  __syncthreads();                 // protect sm reuse across calls
  sm[t] = v;
  __syncthreads();
#pragma unroll
  for (int off = 128; off >= 1; off >>= 1) {
    if (t < off) sm[t] += sm[t + off];
    __syncthreads();
  }
  return sm[0];
}

// --------------------------------------------------------------------------
// Kernel 1: normalize paired rows i and i+N, emit f16 z1 row, sq[i], s_pos[i]
// grid = N blocks of 256 threads, 4 elems/thread (D=1024)
// --------------------------------------------------------------------------
__global__ __launch_bounds__(256) void normalize_kernel(
    const float* __restrict__ z, _Float16* __restrict__ zh,
    float* __restrict__ sq, float* __restrict__ spos, int N) {
  __shared__ float sm[256];
  const int row = blockIdx.x;
  const int t   = threadIdx.x;

  const float4 a = ((const float4*)(z + (size_t)row       * D_DIM))[t];
  const float4 b = ((const float4*)(z + (size_t)(row + N) * D_DIM))[t];

  float ss1 = a.x*a.x + a.y*a.y + a.z*a.z + a.w*a.w;
  float ss2 = b.x*b.x + b.y*b.y + b.z*b.z + b.w*b.w;
  const float S1 = block_reduce_sum(ss1, sm);
  const float S2 = block_reduce_sum(ss2, sm);

  const float inv1 = 1.0f / fmaxf(sqrtf(S1), 1e-12f);
  const float inv2 = 1.0f / fmaxf(sqrtf(S2), 1e-12f);

  const float ax = a.x*inv1, ay = a.y*inv1, az = a.z*inv1, aw = a.w*inv1;
  const float bx = b.x*inv2, by = b.y*inv2, bz = b.z*inv2, bw = b.w*inv2;

  const float dx = ax-bx, dy = ay-by, dz = az-bz, dw = aw-bw;
  const float DP = block_reduce_sum(dx*dx + dy*dy + dz*dz + dw*dw, sm);

  v4h o; o.x = (_Float16)ax; o.y = (_Float16)ay; o.z = (_Float16)az; o.w = (_Float16)aw;
  *(v4h*)(zh + (size_t)row * D_DIM + t*4) = o;

  if (t == 0) {
    sq[row]   = S1 * inv1 * inv1;                       // = sum(z1n^2)
    spos[row] = logf(expf(-DP * SIGMA_INV2S2) + EPS_C) + 1.0f;
  }
}

// --------------------------------------------------------------------------
// Fragment loader for 16-bit 16x32 A-layout (ISA 7.12.2). The B (32x16)
// layout is the same with N in place of M, so for the Gram matrix both
// A- and B-fragments are loaded identically from row blocks of Z1n.
// lane<16:  row = base+lane,    K = k0+0..7  and k0+16..23
// lane>=16: row = base+lane-16, K = k0+8..15 and k0+24..31
// --------------------------------------------------------------------------
__device__ __forceinline__ v16h load_frag(const _Float16* __restrict__ base,
                                          int row0, int k0) {
  const int lane = threadIdx.x & 31;
  const int r    = lane & 15;
  const int ks   = (lane >> 4) << 3;          // 0 or 8
  const _Float16* p = base + (size_t)(row0 + r) * D_DIM + k0 + ks;
  v16h f;
  ((uint4*)&f)[0] = *(const uint4*)(p);       // 8 halves: K = k0+ks .. +7
  ((uint4*)&f)[1] = *(const uint4*)(p + 16);  // 8 halves: K = k0+ks+16 ..
  return f;
}

#define WMMA_F16(A, B, C) \
  __builtin_amdgcn_wmma_f32_16x16x32_f16(false, (A), false, (B), (short)0, (C), false, false)

// --------------------------------------------------------------------------
// Kernel 2: upper-triangular 128x128 C-tiles only (C is symmetric).
// 1D grid of nt*(nt+1)/2 blocks; 8 waves, wave = 32x64 sub-tile.
// Fused epilogue: d2 = max(sq_m + sq_n - 2c, 0); sum exp(-d2/2), skip diag;
// off-diagonal tile-blocks weighted x2.
// --------------------------------------------------------------------------
__global__ __launch_bounds__(256) void gram_kernel(
    const _Float16* __restrict__ zh, const float* __restrict__ sq,
    float* __restrict__ gpart, int nt) {
  // map linear block id -> upper-triangle (bi <= bj)
  int rem = blockIdx.x;
  int bi = 0;
  while (rem >= nt - bi) { rem -= nt - bi; ++bi; }
  const int bj = bi + rem;
  const float wgt = (bi == bj) ? 1.0f : 2.0f;

  const int wave = threadIdx.x >> 5;
  const int lane = threadIdx.x & 31;
  const int m0 = bi * 128 + (wave & 3) * 32;   // 4 wave-rows
  const int n0 = bj * 128 + (wave >> 2) * 64;  // 2 wave-cols

  const v8f vzero = {0.f,0.f,0.f,0.f,0.f,0.f,0.f,0.f};
  v8f acc[2][4];
#pragma unroll
  for (int i = 0; i < 2; ++i)
#pragma unroll
    for (int j = 0; j < 4; ++j) acc[i][j] = vzero;

  for (int k = 0; k < D_DIM; k += 32) {
    const v16h a0 = load_frag(zh, m0,      k);
    const v16h a1 = load_frag(zh, m0 + 16, k);
    const v16h b0 = load_frag(zh, n0,      k);
    const v16h b1 = load_frag(zh, n0 + 16, k);
    const v16h b2 = load_frag(zh, n0 + 32, k);
    const v16h b3 = load_frag(zh, n0 + 48, k);
    acc[0][0] = WMMA_F16(a0, b0, acc[0][0]);
    acc[0][1] = WMMA_F16(a0, b1, acc[0][1]);
    acc[0][2] = WMMA_F16(a0, b2, acc[0][2]);
    acc[0][3] = WMMA_F16(a0, b3, acc[0][3]);
    acc[1][0] = WMMA_F16(a1, b0, acc[1][0]);
    acc[1][1] = WMMA_F16(a1, b1, acc[1][1]);
    acc[1][2] = WMMA_F16(a1, b2, acc[1][2]);
    acc[1][3] = WMMA_F16(a1, b3, acc[1][3]);
  }

  // C layout: VGPR r, lanes 0-15 -> M=r, N=lane; lanes 16-31 -> M=r+8.
  const int nl = lane & 15;
  const int mh = (lane >> 4) << 3;
  float gsum = 0.0f;
#pragma unroll
  for (int i = 0; i < 2; ++i) {
#pragma unroll
    for (int j = 0; j < 4; ++j) {
      const int mbase = m0 + i * 16 + mh;
      const int n     = n0 + j * 16 + nl;
      const float sqn = sq[n];
#pragma unroll
      for (int r = 0; r < 8; ++r) {
        const int m   = mbase + r;
        const float c = acc[i][j][r];
        const float d2 = fmaxf(sq[m] + sqn - 2.0f * c, 0.0f);
        const float g  = __expf(-d2 * SIGMA_INV2S2);
        if (m != n) gsum += g;
      }
    }
  }

  __shared__ float sm[256];
  const float G = block_reduce_sum(gsum, sm);
  if (threadIdx.x == 0) gpart[blockIdx.x] = G * wgt;
}

// --------------------------------------------------------------------------
// Kernel 3: deterministic final reduction + loss
// --------------------------------------------------------------------------
__global__ __launch_bounds__(256) void finalize_kernel(
    const float* __restrict__ spos, const float* __restrict__ gpart,
    float* __restrict__ out, int N, int nblocks) {
  __shared__ float sm[256];
  const int t = threadIdx.x;

  float s = 0.0f;
  for (int i = t; i < N; i += 256) s += spos[i];
  const float S = block_reduce_sum(s, sm);

  float g = 0.0f;
  for (int i = t; i < nblocks; i += 256) g += gpart[i];
  const float G = block_reduce_sum(g, sm);

  if (t == 0) {
    const float star_mean = G / ((float)N * (float)(N - 1)) + EPS_C;
    out[0] = -(S / (float)N) + ALPHA_C * star_mean;
  }
}

// --------------------------------------------------------------------------
extern "C" void kernel_launch(void* const* d_in, const int* in_sizes, int n_in,
                              void* d_out, int out_size, void* d_ws, size_t ws_size,
                              hipStream_t stream) {
  const float* z = (const float*)d_in[0];
  const int rows = in_sizes[0] / D_DIM;   // 8192
  const int N    = rows / 2;              // 4096

  // workspace layout
  char* ws = (char*)d_ws;
  _Float16* zh  = (_Float16*)ws;                       // N*D f16   (8 MB)
  size_t off = (size_t)N * D_DIM * sizeof(_Float16);
  float* sq    = (float*)(ws + off); off += (size_t)N * sizeof(float);
  float* spos  = (float*)(ws + off); off += (size_t)N * sizeof(float);
  float* gpart = (float*)(ws + off);

  const int nt   = N / 128;               // 32 tile-blocks per dim
  const int ntri = nt * (nt + 1) / 2;     // 528 upper-tri blocks

  normalize_kernel<<<N, 256, 0, stream>>>(z, zh, sq, spos, N);

  gram_kernel<<<ntri, 256, 0, stream>>>(zh, sq, gpart, nt);

  finalize_kernel<<<1, 256, 0, stream>>>(spos, gpart, (float*)d_out, N, ntri);
}
